// QuantumLSTMCell_81003083202960
// MI455X (gfx1250) — matrix operand order repaired and checked by reading, not compiled, not run on hardware
//
#include <hip/hip_runtime.h>
#include <hip/hip_bf16.h>

// QuantumLSTM cell, fully fused for MI455X (gfx1250, wave32).
// B=65536, IN=HID=512, fan1=1024, NGATES=4, NQ=8.
// Memory-bound (~640MB @ 23.3TB/s => ~27us floor): stream x,h,c once,
// keep W1/W2/b1/b2 resident in LDS, do both einsums with v_wmma_f32_16x16x32_f16.
// Double-buffered LDS staging: one barrier per K-chunk, global loads for
// chunk k+1 issued before the WMMAs of chunk k.

typedef _Float16 v16h __attribute__((ext_vector_type(16)));
typedef _Float16 h8   __attribute__((ext_vector_type(8)));
typedef _Float16 h4   __attribute__((ext_vector_type(4)));
typedef float    v8f  __attribute__((ext_vector_type(8)));

#define BATCH   65536
#define HID     512
#define FAN1    1024
#define NGATES  4
#define NQ      8
#define ROWS_PER_BLOCK 128
#define WAVES   8
#define NKC     (FAN1 / 32)   // 32 K-chunks of width 32

__device__ __forceinline__ float fast_sigmoid(float x) {
    return 1.0f / (1.0f + __expf(-x));
}
__device__ __forceinline__ float fast_tanh(float x) {
    float ax = fabsf(x);
    float t  = __expf(-2.0f * ax);
    float r  = (1.0f - t) / (1.0f + t);
    return copysignf(r, x);
}

union BFrag {
    v16h v;
    h8   h[2];
};

__global__ __launch_bounds__(256, 1)
void qlstm_fused_kernel(const float* __restrict__ x,
                        const float* __restrict__ hprev,
                        const float* __restrict__ cprev,
                        const float* __restrict__ W1,
                        const float* __restrict__ b1,
                        const float* __restrict__ W2,
                        const float* __restrict__ b2,
                        float* __restrict__ h_out,
                        float* __restrict__ c_out) {
    // W1 pre-swizzled into WMMA B-fragments: [kc][ntile][lane][j] f16  -> 64KB
    __shared__ __align__(32) _Float16 sW1[NKC * 2 * 32 * 16];
    // W2 compact f16 [g][hid][q] -> 32KB
    __shared__ __align__(16) _Float16 sW2[NGATES * HID * NQ];
    __shared__ float sB2[NGATES * HID];   // 8KB
    __shared__ float sB1[32];
    // double-buffered staged A chunk, pre-swizzled: [buf][wave][lane][j] -> 16KB
    __shared__ __align__(32) _Float16 sA[2][WAVES * 32 * 16];
    // z transpose buffer: [wave][m][k] f16 -> 8KB
    __shared__ __align__(16) _Float16 sZ[WAVES * 16 * 32];

    const int t    = threadIdx.x;
    const int wave = t >> 5;
    const int lane = t & 31;
    const int laneM = lane & 15;    // row (A) or col (B/C) within fragment
    const int hs    = lane >> 4;    // half-select
    const int r0    = blockIdx.x * ROWS_PER_BLOCK;

    // ---- one-time weight staging into LDS ----
    for (int i = t; i < NGATES * HID * NQ; i += 256) sW2[i] = (_Float16)W2[i];
    for (int i = t; i < NGATES * HID; i += 256)      sB2[i] = b2[i];
    if (t < 32) sB1[t] = b1[t];
    // W1 B-fragment layout: lane L -> col n=L%16, K = j + 16*(L/16) within chunk
    for (int F = t; F < NKC * 2 * 32; F += 256) {
        int kc   = F >> 6;
        int nt   = (F >> 5) & 1;
        int ln   = F & 31;
        int n    = nt * 16 + (ln & 15);
        int kofs = kc * 32 + 16 * (ln >> 4);
        #pragma unroll
        for (int j = 0; j < 16; ++j)
            sW1[F * 16 + j] = (_Float16)W1[n * FAN1 + kofs + j];
    }

    // ---- per-thread staging geometry (invariant across K-chunks) ----
    const int rowl = t >> 3;              // 0..31 (row within 32-row pass)
    const int kl   = (t & 7) * 4;         // 0,4,...,28 within 32-wide chunk
    const int hsl  = kl >> 4;
    const int jj   = kl & 15;
    // store index for pass p is storeBase + p*1024 halves (p covers rows +32p)
    const int storeBase = (((rowl >> 4) * 32) + (rowl & 15) + 16 * hsl) * 16 + jj;
    const size_t gbase  = (size_t)(r0 + rowl) * 512 + kl;

    float4 vv[4];
    auto loadChunk = [&](int kc) {
        const int kbase = kc * 32;
        const float* src = (kbase < 512) ? x : hprev;
        const int koff   = kbase & 511;
        #pragma unroll
        for (int p = 0; p < 4; ++p)
            vv[p] = *(const float4*)(src + gbase + (size_t)p * 32 * 512 + koff);
    };
    auto storeChunk = [&](int buf) {
        #pragma unroll
        for (int p = 0; p < 4; ++p) {
            h4 hv = { (_Float16)vv[p].x, (_Float16)vv[p].y,
                      (_Float16)vv[p].z, (_Float16)vv[p].w };
            *(h4*)&sA[buf][storeBase + p * 1024] = hv;
        }
    };

    // ---- GEMM1: z_pre = combined(16x1024) x W1^T(1024x32), per-wave 16-row tile ----
    v8f acc0 = {};   // cols 0..15  (gates 0,1)
    v8f acc1 = {};   // cols 16..31 (gates 2,3)

    loadChunk(0);
    storeChunk(0);
    __syncthreads();   // also covers the weight staging above

    for (int kc = 0; kc < NKC; ++kc) {
        const int buf = kc & 1;
        if (kc + 1 < NKC) loadChunk(kc + 1);   // overlap next chunk's loads

        v16h aF  = *(const v16h*)&sA[buf][(wave * 32 + lane) * 16];
        v16h bF0 = *(const v16h*)&sW1[((kc * 2 + 0) * 32 + lane) * 16];
        v16h bF1 = *(const v16h*)&sW1[((kc * 2 + 1) * 32 + lane) * 16];
        acc0 = __builtin_amdgcn_wmma_f32_16x16x32_f16(false, aF, false, bF0,
                                                      (short)0, acc0, false, false);
        acc1 = __builtin_amdgcn_wmma_f32_16x16x32_f16(false, aF, false, bF1,
                                                      (short)0, acc1, false, false);

        if (kc + 1 < NKC) storeChunk(buf ^ 1);
        __syncthreads();
    }

    // ---- z = tanh(z_pre + b1), transpose C-layout -> A-layout via LDS ----
    {
        const float bb0 = sB1[laneM];
        const float bb1 = sB1[16 + laneM];
        #pragma unroll
        for (int j = 0; j < 8; ++j) {
            const int m = j + hs * 8;
            sZ[(wave * 16 + m) * 32 + laneM]      = (_Float16)fast_tanh(acc0[j] + bb0);
            sZ[(wave * 16 + m) * 32 + 16 + laneM] = (_Float16)fast_tanh(acc1[j] + bb1);
        }
    }
    __syncthreads();

    v16h zF;
    {
        const _Float16* zrow = &sZ[(wave * 16 + laneM) * 32];
        h8 lo = *(const h8*)(zrow + hs * 16);       // K = hs*16 + 0..7
        h8 hi = *(const h8*)(zrow + hs * 16 + 8);   // K = hs*16 + 8 + 0..7
        #pragma unroll
        for (int j = 0; j < 8; ++j) { zF[j] = lo[j]; zF[8 + j] = hi[j]; }
    }

    // ---- GEMM2 (block-diagonal) + fused LSTM elementwise ----
    // Output tile: rows = same 16 batch rows, cols = 16 hidden per gate.
    for (int ct = 0; ct < HID / 16; ++ct) {
        v8f ga[4];
        #pragma unroll
        for (int g = 0; g < 4; ++g) {
            BFrag bf;
            bf.h[0] = (h8){};
            bf.h[1] = (h8){};
            // lane carries K = [16*hs, 16*hs+16); gate g occupies K = [8g, 8g+8)
            // -> EXEC-masked b128 load straight into one half of the fragment
            if ((g >> 1) == hs) {
                bf.h[g & 1] = *(const h8*)&sW2[((g * HID) + ct * 16 + laneM) * NQ];
            }
            v8f a = {};
            a = __builtin_amdgcn_wmma_f32_16x16x32_f16(false, zF, false, bf.v,
                                                       (short)0, a, false, false);
            ga[g] = a;
        }
        const int hc = ct * 16 + laneM;
        const float bi = sB2[0 * HID + hc];
        const float bf_ = sB2[1 * HID + hc];
        const float bo = sB2[2 * HID + hc];
        const float bg = sB2[3 * HID + hc];
        #pragma unroll
        for (int j = 0; j < 8; ++j) {
            const int brow = r0 + wave * 16 + j + hs * 8;
            const size_t idx = (size_t)brow * HID + hc;
            const float iv = fast_sigmoid(ga[0][j] + bi);
            const float fv = fast_sigmoid(ga[1][j] + bf_);
            const float ov = fast_sigmoid(ga[2][j] + bo);
            const float gv = fast_sigmoid(ga[3][j] + bg);
            const float cv = cprev[idx];
            const float cn = fv * cv + iv * gv;
            const float hn = ov * fast_tanh(cn);
            h_out[idx] = hn;
            c_out[idx] = cn;
        }
    }
}

extern "C" void kernel_launch(void* const* d_in, const int* in_sizes, int n_in,
                              void* d_out, int out_size, void* d_ws, size_t ws_size,
                              hipStream_t stream) {
    const float* x  = (const float*)d_in[0];
    const float* h  = (const float*)d_in[1];
    const float* c  = (const float*)d_in[2];
    const float* W1 = (const float*)d_in[3];
    const float* b1 = (const float*)d_in[4];
    const float* W2 = (const float*)d_in[5];
    const float* b2 = (const float*)d_in[6];
    float* out = (float*)d_out;
    float* h_new = out;                              // first output
    float* c_new = out + (size_t)BATCH * HID;        // second output

    dim3 grid(BATCH / ROWS_PER_BLOCK);  // 512 blocks
    dim3 block(256);                    // 8 waves (wave32)
    qlstm_fused_kernel<<<grid, block, 0, stream>>>(x, h, c, W1, b1, W2, b2,
                                                   h_new, c_new);
}
